// RowParallelLinearWithDelta_28973849379102
// MI455X (gfx1250) — matrix operand order repaired and checked by reading, not compiled, not run on hardware
//
#include <hip/hip_runtime.h>

typedef __bf16 v16bf __attribute__((ext_vector_type(16)));
typedef float v8f __attribute__((ext_vector_type(8)));

#define TILE_M 256
#define TILE_N 128
#define TILE_K 32
#define XSTR 40   // ushort row stride (32 K + 8 pad) -> conflict-free b128 reads
#define PERM_BYTES 16384

union FragU {
  uint4 q[2];
  v16bf v;
};

// Truncation split: f == hi + lo to ~2^-16 relative (lo holds the next 8 bits).
__device__ __forceinline__ void split2(float f, unsigned& h, unsigned& l) {
  unsigned u = __float_as_uint(f);
  h = u >> 16;
  float r = f - __uint_as_float(u & 0xFFFF0000u);
  l = __float_as_uint(r) >> 16;
}

__device__ __forceinline__ void split4(float4 v, uint2& ph, uint2& pl) {
  unsigned h0, l0, h1, l1, h2, l2, h3, l3;
  split2(v.x, h0, l0);
  split2(v.y, h1, l1);
  split2(v.z, h2, l2);
  split2(v.w, h3, l3);
  ph = make_uint2(h0 | (h1 << 16), h2 | (h3 << 16));
  pl = make_uint2(l0 | (l1 << 16), l2 | (l3 << 16));
}

// A fragment (16x32 bf16): lane holds K runs [akoff, akoff+8) and [akoff+16, +8)
__device__ __forceinline__ v16bf load_frag_a(const unsigned short* p) {
  FragU f;
  f.q[0] = *(const uint4*)p;
  f.q[1] = *(const uint4*)(p + 16);
  return f.v;
}
// B fragment (32x16 bf16): lane holds K runs [bkoff, bkoff+8) and [bkoff+16, +8)
__device__ __forceinline__ v16bf load_frag_b(const unsigned short* p) {
  FragU f;
  f.q[0] = *(const uint4*)p;
  f.q[1] = *(const uint4*)(p + 16);
  return f.v;
}

// ---------------------------------------------------------------------------
// Kernel 0: pre-split X into packed bf16 hi/lo planes (done once per call).
// ---------------------------------------------------------------------------
__global__ void prepack_x_kernel(const float* __restrict__ x,
                                 unsigned short* __restrict__ xh,
                                 unsigned short* __restrict__ xl, int n4) {
  int i = blockIdx.x * blockDim.x + threadIdx.x;
  if (i >= n4) return;
  float4 v = ((const float4*)x)[i];
  uint2 ph, pl;
  split4(v, ph, pl);
  ((uint2*)xh)[i] = ph;
  ((uint2*)xl)[i] = pl;
}

// ---------------------------------------------------------------------------
// Kernel 1: bucket tokens by delta index.
// ws ints: [0..3]=counts [4..7]=offsets [8..11]=cursor [12..]=perm
// ---------------------------------------------------------------------------
__global__ void group_tokens_kernel(const int* __restrict__ indices, int T,
                                    int* __restrict__ ws) {
  const int tid = threadIdx.x;
  if (tid < 12) ws[tid] = 0;
  __syncthreads();
  int* counts = ws;
  int* offsets = ws + 4;
  int* cursor = ws + 8;
  int* perm = ws + 12;
  for (int t = tid; t < T; t += blockDim.x) atomicAdd(&counts[indices[t] & 3], 1);
  __syncthreads();
  if (tid == 0) {
    int acc = 0;
    for (int d = 0; d < 4; ++d) {
      offsets[d] = acc;
      cursor[d] = acc;
      acc += counts[d];
    }
  }
  __syncthreads();
  for (int t = tid; t < T; t += blockDim.x) {
    int d = indices[t] & 3;
    perm[atomicAdd(&cursor[d], 1)] = t;  // bucket order irrelevant to output
  }
}

// ---------------------------------------------------------------------------
// Kernel 2: fused bf16x3 GEMM  out[perm[r]] = x[perm[r]] @ (baseW + dW[d])^T
// Block 256x128, 8 waves of 64x64 (4x4 WMMA16 tiles), K-slab 32.
// acc += Ah*Bh + Ah*Bl + Al*Bh  (fp32 accumulate, ~1e-5 rel err vs fp32)
// ---------------------------------------------------------------------------
template <bool PREPACK>
__global__ __launch_bounds__(256) void fused_delta_gemm_kernel(
    const float* __restrict__ x, const unsigned short* __restrict__ xh,
    const unsigned short* __restrict__ xl, const float* __restrict__ bw,
    const int* __restrict__ qw, const int* __restrict__ qz,
    const float* __restrict__ sc, const int* __restrict__ ws,
    float* __restrict__ out, int T, int IN, int OUT) {
  const int d = blockIdx.z;
  const int cnt = ws[d];
  const int off = ws[4 + d];
  const int* perm = ws + 12;

  const int row0 = blockIdx.y * TILE_M;
  if (row0 >= cnt) return;  // block-uniform exit
  const int n0 = blockIdx.x * TILE_N;

  __shared__ __align__(16) unsigned short lxh[TILE_M][XSTR];
  __shared__ __align__(16) unsigned short lxl[TILE_M][XSTR];
  __shared__ __align__(16) unsigned short lwh[TILE_N][XSTR];
  __shared__ __align__(16) unsigned short lwl[TILE_N][XSTR];
  __shared__ int ltok[TILE_M];
  __shared__ float2 lzs[TILE_N];  // x = zero-point, y = scale per out-row

  const int tid = threadIdx.x;
  {
    int r = row0 + tid;
    ltok[tid] = (r < cnt) ? perm[off + r] : -1;
    if (tid < TILE_N) {
      int o = n0 + tid;
      int zp = qz[d * (OUT >> 3) + (o >> 3)];
      lzs[tid] = make_float2((float)((zp >> ((o & 7) * 4)) & 0xF),
                             sc[d * OUT + o]);
    }
  }
  __syncthreads();

  const int lane = tid & 31;
  const int wave = tid >> 5;
  const int mbase = (wave >> 1) * 64;  // 4 wave-rows
  const int nbase = (wave & 1) * 64;   // 2 wave-cols
  const int lrow = lane & 15;
  const int akoff = (lane >> 4) * 8;   // A: K runs {0..7,16..23} / {8..15,24..31}
  const int bkoff = (lane >> 4) * 8;   // B: K runs {0..7,16..23} / {8..15,24..31}

  v8f acc[4][4];
  const v8f vzero = {0.f, 0.f, 0.f, 0.f, 0.f, 0.f, 0.f, 0.f};
#pragma unroll
  for (int mt = 0; mt < 4; ++mt)
#pragma unroll
    for (int nt = 0; nt < 4; ++nt) acc[mt][nt] = vzero;

  const int ldq = IN >> 3;
  const int kblocks = IN / TILE_K;
  const int srow = tid >> 3;  // 0..31: 32 rows per staging pass
  const int sc8 = tid & 7;    // 8 threads per row, 4 elems each

#pragma unroll 1
  for (int kb = 0; kb < kblocks; ++kb) {
    const int kbase = kb * TILE_K;
    const int kcol = kbase + sc8 * 4;

    // ---- stage gathered X slab (256 rows x 32 K), hi/lo planes ----
#pragma unroll
    for (int i = 0; i < 8; ++i) {
      int r = srow + i * 32;
      int t = ltok[r];
      uint2 ph = make_uint2(0u, 0u), pl = make_uint2(0u, 0u);
      if (t >= 0) {
        if (PREPACK) {
          size_t base = (size_t)t * IN + kcol;
          ph = *(const uint2*)&xh[base];
          pl = *(const uint2*)&xl[base];
        } else {
          float4 v = *(const float4*)&x[(size_t)t * IN + kcol];
          split4(v, ph, pl);
        }
      }
      *(uint2*)&lxh[r][sc8 * 4] = ph;
      *(uint2*)&lxl[r][sc8 * 4] = pl;
    }

    // ---- stage fused weight slab (128 x 32): base + scale*(nib - zero) ----
#pragma unroll
    for (int i = 0; i < 4; ++i) {
      int r = srow + i * 32;
      int o = n0 + r;
      float4 b = *(const float4*)&bw[(size_t)o * IN + kcol];
      int q = qw[((size_t)d * OUT + o) * ldq + (kcol >> 3)];
      float2 zs = lzs[r];
      int sh = (kcol & 7) * 4;  // 0 or 16
      b.x += zs.y * ((float)((q >> (sh + 0)) & 0xF) - zs.x);
      b.y += zs.y * ((float)((q >> (sh + 4)) & 0xF) - zs.x);
      b.z += zs.y * ((float)((q >> (sh + 8)) & 0xF) - zs.x);
      b.w += zs.y * ((float)((q >> (sh + 12)) & 0xF) - zs.x);
      uint2 ph, pl;
      split4(b, ph, pl);
      *(uint2*)&lwh[r][sc8 * 4] = ph;
      *(uint2*)&lwl[r][sc8 * 4] = pl;
    }
    __syncthreads();

    // ---- bf16x3 WMMA: 48 v_wmma_f32_16x16x32_bf16 per wave per slab ----
    v16bf ah[4], al[4];
#pragma unroll
    for (int mt = 0; mt < 4; ++mt) {
      const unsigned short* pr = &lxh[mbase + mt * 16 + lrow][akoff];
      const unsigned short* qr = &lxl[mbase + mt * 16 + lrow][akoff];
      ah[mt] = load_frag_a(pr);
      al[mt] = load_frag_a(qr);
    }
#pragma unroll
    for (int nt = 0; nt < 4; ++nt) {
      v16bf bh = load_frag_b(&lwh[nbase + nt * 16 + lrow][bkoff]);
      v16bf bl = load_frag_b(&lwl[nbase + nt * 16 + lrow][bkoff]);
#pragma unroll
      for (int mt = 0; mt < 4; ++mt) {
        acc[mt][nt] = __builtin_amdgcn_wmma_f32_16x16x32_bf16(
            false, ah[mt], false, bh, (short)0, acc[mt][nt], false, false);
        acc[mt][nt] = __builtin_amdgcn_wmma_f32_16x16x32_bf16(
            false, ah[mt], false, bl, (short)0, acc[mt][nt], false, false);
        acc[mt][nt] = __builtin_amdgcn_wmma_f32_16x16x32_bf16(
            false, al[mt], false, bh, (short)0, acc[mt][nt], false, false);
      }
    }
    __syncthreads();
  }

  // ---- scatter through the permutation (C/D: lanes16-31 hold M=r+8) ----
  const int mhalf = (lane >> 4) * 8;
#pragma unroll
  for (int mt = 0; mt < 4; ++mt) {
#pragma unroll
    for (int nt = 0; nt < 4; ++nt) {
#pragma unroll
      for (int r = 0; r < 8; ++r) {
        int ml = mbase + mt * 16 + mhalf + r;
        int t = ltok[ml];
        if (t >= 0)
          out[(size_t)t * OUT + n0 + nbase + nt * 16 + (lane & 15)] =
              acc[mt][nt][r];
      }
    }
  }
}

// ---------------------------------------------------------------------------
extern "C" void kernel_launch(void* const* d_in, const int* in_sizes, int n_in,
                              void* d_out, int out_size, void* d_ws,
                              size_t ws_size, hipStream_t stream) {
  const float* x = (const float*)d_in[0];
  const float* bw = (const float*)d_in[1];
  const int* qw = (const int*)d_in[2];
  const int* qz = (const int*)d_in[3];
  const float* sc = (const float*)d_in[4];
  const int* indices = (const int*)d_in[5];
  float* out = (float*)d_out;
  int* ws = (int*)d_ws;

  const int T = in_sizes[5];         // 2048
  const int IN = in_sizes[0] / T;    // 4096
  const int OUT = in_sizes[1] / IN;  // 4096
  const int D = in_sizes[4] / OUT;   // 4

  group_tokens_kernel<<<1, 256, 0, stream>>>(indices, T, ws);

  const size_t xplane = (size_t)T * IN * sizeof(unsigned short);
  const bool prepack = ws_size >= PERM_BYTES + 2 * xplane;

  dim3 grid((OUT + TILE_N - 1) / TILE_N, (T + TILE_M - 1) / TILE_M, D);
  if (prepack) {
    unsigned short* xh = (unsigned short*)((char*)d_ws + PERM_BYTES);
    unsigned short* xl = xh + (size_t)T * IN;
    int n4 = (T * IN) / 4;
    prepack_x_kernel<<<(n4 + 255) / 256, 256, 0, stream>>>(x, xh, xl, n4);
    fused_delta_gemm_kernel<true><<<grid, 256, 0, stream>>>(
        x, xh, xl, bw, qw, qz, sc, ws, out, T, IN, OUT);
  } else {
    fused_delta_gemm_kernel<false><<<grid, 256, 0, stream>>>(
        x, nullptr, nullptr, bw, qw, qz, sc, ws, out, T, IN, OUT);
  }
}
